// C_RNN_24996709662948
// MI455X (gfx1250) — matrix operand order repaired and checked by reading, not compiled, not run on hardware
//
#include <hip/hip_runtime.h>
#include <math.h>

#define T_FRAMES 8192
#define D_IN     106
#define HDIM     512
#define G4       2048
#define KPAD     256
#define PHONE    48

typedef __attribute__((ext_vector_type(16))) __bf16       v16bf;
typedef __attribute__((ext_vector_type(8)))  float        v8f;
typedef __attribute__((ext_vector_type(4)))  unsigned int uv4;
typedef __attribute__((ext_vector_type(4)))  unsigned int v4u;
typedef __attribute__((ext_vector_type(8)))  int          v8i;
typedef __attribute__((ext_vector_type(4)))  int          v4i;

union FragU { uv4 q[2]; v16bf v; };

// round-to-nearest-even f32 -> bf16 (bit pattern in u16)
__device__ inline unsigned short f2bf(float f) {
  unsigned u = __float_as_uint(f);
  u += 0x7FFFu + ((u >> 16) & 1u);
  return (unsigned short)(u >> 16);
}

// low 32 bits of a generic pointer into LDS == byte offset in the block's allocation
__device__ inline unsigned lds_offset(const void* p) {
  return (unsigned)(unsigned long long)p;
}

// ---- TDM descriptor build (ISA 8.3/8.4), 2-D tile where tile == tensor --------------
// group0: [1:0]=count=1, [63:32]=lds_addr, [120:64]=global_addr, [127:126]=type=2
__device__ inline v4u tdm_group0(unsigned lds_addr, unsigned long long gaddr) {
  v4u g;
  g[0] = 1u;
  g[1] = lds_addr;
  g[2] = (unsigned)(gaddr & 0xFFFFFFFFull);
  g[3] = (unsigned)((gaddr >> 32) & 0x01FFFFFFull) | (2u << 30);
  return g;
}
// group1: [17:16]=data_size(1->2B), [79:48]=tensor_dim0, [111:80]=tensor_dim1,
//         [127:112]=tile_dim0, [143:128]=tile_dim1, [207:160]=tensor_dim0_stride
__device__ inline v8i tdm_group1_2d(unsigned dim0, unsigned rows, unsigned stride0) {
  v8i g;
  g[0] = (int)(1u << 16);                        // data_size = 2 bytes
  g[1] = (int)(dim0 << 16);                      // tensor_dim0[15:0]
  g[2] = (int)((dim0 >> 16) | (rows << 16));     // tensor_dim0[31:16] | tensor_dim1[15:0]
  g[3] = (int)((rows >> 16) | (dim0 << 16));     // tensor_dim1[31:16] | tile_dim0
  g[4] = (int)rows;                              // tile_dim1 (tile_dim2 = 0)
  g[5] = (int)stride0;                           // tensor_dim0_stride[31:0]
  g[6] = 0;                                      // stride0[47:32] | dim1_stride[15:0]
  g[7] = 0;
  return g;
}

// ---- async global -> LDS 16B copy (GV mode: VDST = LDS byte offset) -----------------
__device__ inline void async_ld_b128(unsigned lds_off, const void* gptr) {
  asm volatile("global_load_async_to_lds_b128 %0, %1, off"
               :: "v"(lds_off), "v"(gptr) : "memory");
}
__device__ inline void wait_asynccnt0() {
  asm volatile("s_wait_asynccnt 0x0" ::: "memory");
}

// A-matrix 16x32 bf16 fragment: lanes 0-15 -> M=lane, K=0..15;
// lanes 16-31 -> M=lane-16, K=16..31. Source row-major, ldk elems/row.
__device__ inline v16bf load_a_frag(const unsigned short* base, int ldk) {
  int lane = threadIdx.x & 31;
  int m    = lane & 15;
  int kb   = (lane >> 4) << 4;                 // 0 or 16
  const unsigned short* p = base + (long)m * ldk + kb;
  FragU u;
  u.q[0] = *(const uv4*)(p);                   // K = kb .. kb+7
  u.q[1] = *(const uv4*)(p + 8);               // K = kb+8 .. kb+15
  return u.v;
}

// B-matrix 32x16 bf16 fragment: lane = column N (mod 16); lanes 0-15 hold K=0..15,
// lanes 16-31 hold K=16..31. Source W stored [N][K] row-major (W[n][k] == B(k,n)).
__device__ inline v16bf load_b_frag(const unsigned short* wbase, int ldk) {
  int lane = threadIdx.x & 31;
  int n    = lane & 15;
  int kb   = (lane >> 4) << 4;                 // 0 or 16
  const unsigned short* p = wbase + (long)n * ldk + kb;
  FragU u;
  u.q[0] = *(const uv4*)(p);
  u.q[1] = *(const uv4*)(p + 8);
  return u.v;
}

// B fragment with every column = h (matvec through the matrix pipe)
__device__ inline v16bf load_b_frag_bcast(const unsigned short* hvec) {
  int lane = threadIdx.x & 31;
  int kb   = (lane >> 4) << 4;
  const unsigned short* p = hvec + kb;
  FragU u;
  u.q[0] = *(const uv4*)(p);
  u.q[1] = *(const uv4*)(p + 8);
  return u.v;
}

// ---------------- prep: bf16 weight copies, fused bias, h0 staging, sync reset --------
__global__ void prep_kernel(const float* __restrict__ w_ih, const float* __restrict__ w_hh,
                            const float* __restrict__ out_w, const float* __restrict__ b_ih,
                            const float* __restrict__ b_hh, const float* __restrict__ h0,
                            unsigned short* wih_bf, unsigned short* whh_bf,
                            unsigned short* outw_bf, float* bias, float* hbuf,
                            unsigned* sync) {
  long idx = (long)blockIdx.x * blockDim.x + threadIdx.x;
  const long N0 = (long)G4 * KPAD;
  const long N1 = N0 + (long)G4 * HDIM;
  const long N2 = N1 + (long)PHONE * HDIM;
  const long N3 = N2 + G4;
  const long N4 = N3 + HDIM;
  if (idx < N0) {
    int n = (int)(idx / KPAD), k = (int)(idx % KPAD);
    wih_bf[idx] = (k < 249) ? f2bf(w_ih[(long)n * 249 + k]) : (unsigned short)0;
  } else if (idx < N1) {
    long i = idx - N0; whh_bf[i] = f2bf(w_hh[i]);
  } else if (idx < N2) {
    long i = idx - N1; outw_bf[i] = f2bf(out_w[i]);
  } else if (idx < N3) {
    int i = (int)(idx - N2); bias[i] = b_ih[i] + b_hh[i];
  } else if (idx < N4) {
    int i = (int)(idx - N3); hbuf[i] = h0[i];           // buffer 0 = h at t=0
  } else if (idx == N4) {
    *sync = 0u;
  }
}

// ---------------- conv(9x5) + maxpool(1x3) + concat MFCC -> bf16 [T, 256] ------------
__global__ void feat_kernel(const float* __restrict__ in, const float* __restrict__ conv_w,
                            const float* __restrict__ conv_b, unsigned short* xin) {
  __shared__ float cw[450];
  __shared__ float cb[10];
  int t = blockIdx.x;
  int j = threadIdx.x;
  if (j < 450) cw[j] = conv_w[j];
  if (j < 10)  cb[j] = conv_b[j];
  __syncthreads();
  unsigned short val;
  if (j < 210) {
    int f = j / 21, pw = j % 21;
    float mx = -1e30f;
    for (int w3 = 0; w3 < 3; ++w3) {
      int w = pw * 3 + w3;
      float s = cb[f];
      for (int dh = 0; dh < 9; ++dh) {
        int tr = t + dh - 4;                              // edge-replicate == clamp
        tr = tr < 0 ? 0 : (tr > T_FRAMES - 1 ? T_FRAMES - 1 : tr);
        const float* xr = in + (long)tr * D_IN + 39 + w;
        const float* wr = cw + f * 45 + dh * 5;
        for (int dw = 0; dw < 5; ++dw) s += xr[dw] * wr[dw];
      }
      mx = fmaxf(mx, s);
    }
    val = f2bf(mx);
  } else if (j < 249) {
    val = f2bf(in[(long)t * D_IN + (j - 210)]);           // MFCC passthrough
  } else {
    val = 0;                                              // K padding
  }
  xin[(long)t * KPAD + j] = val;
}

// ---------------- gates GEMM: TDM-staged B tile in LDS, 8 M-tiles per block ----------
__global__ void gates_gemm_kernel(const unsigned short* __restrict__ xin,
                                  const unsigned short* __restrict__ wih,
                                  const float* __restrict__ bias, float* gates) {
  __shared__ unsigned short btile[16 * KPAD] __attribute__((aligned(128)));  // 8 KB
  int wave = threadIdx.x >> 5;
  int lane = threadIdx.x & 31;
  int mg = blockIdx.x >> 7;                  // 0..63 -> 8 M-tiles each
  int nt = blockIdx.x & 127;                 // shared N-tile
  int m0 = (mg * 8 + wave) * 16;
  int n0 = nt * 16;

  if (wave == 0) {                           // one TDM descriptor stages W_ih[n0:n0+16, 0:256]
    v4u g0 = tdm_group0(lds_offset(&btile[0]),
                        (unsigned long long)(wih + (long)n0 * KPAD));
    v8i g1 = tdm_group1_2d(KPAD, 16, KPAD);
    v4i z4 = {0, 0, 0, 0};
    v8i z8 = {0, 0, 0, 0, 0, 0, 0, 0};
    __builtin_amdgcn_tensor_load_to_lds(g0, g1, z4, z4, z8, 0);
    __builtin_amdgcn_s_wait_tensorcnt(0);
  }
  __syncthreads();

  v8f acc = {};
  const unsigned short* abase = xin + (long)m0 * KPAD;
  for (int k0 = 0; k0 < KPAD; k0 += 32) {
    __builtin_prefetch(abase + k0 + 64, 0, 1);
    v16bf a = load_a_frag(abase + k0, KPAD);
    v16bf b = load_b_frag(btile + k0, KPAD);
    acc = __builtin_amdgcn_wmma_f32_16x16x32_bf16(false, a, false, b, (short)0, acc,
                                                  false, false);
  }
  int col   = n0 + (lane & 15);
  int rbase = m0 + ((lane >> 4) << 3);
  float bv = bias[col];
  for (int r = 0; r < 8; ++r)
    gates[(long)(rbase + r) * G4 + col] = acc[r] + bv;
}

// ---------------- persistent LSTM: 16 blocks, WMMA matvec + atomic grid barrier ------
__global__ void lstm_kernel(const float* __restrict__ gates,
                            const unsigned short* __restrict__ whh,
                            const float* __restrict__ c0, float* hbuf,
                            unsigned short* hs_bf, unsigned* sync) {
  __shared__ unsigned short h_lds[HDIM] __attribute__((aligned(16)));
  __shared__ float r_lds[128];
  int tid  = threadIdx.x;
  int lane = tid & 31;
  int wave = tid >> 5;
  int b    = blockIdx.x;                     // 0..15, owns h[j], j in [32b, 32b+32)
  int jbase = b * 32;
  // wave w handles 16 gate rows: {i,i,f,f,g,g,o,o} x 16-row subtiles
  int rowbase = ((wave >> 1) * 512) + jbase + ((wave & 1) << 4);
  const unsigned short* arows = whh + (long)rowbase * HDIM;
  float c_reg = (tid < 32) ? c0[jbase + tid] : 0.f;

  for (int t = 0; t < T_FRAMES; ++t) {
    const float* hin = hbuf + (t & 1) * HDIM;           // double-buffered h
    float*      hout = hbuf + ((t + 1) & 1) * HDIM;
    h_lds[tid]       = f2bf(hin[tid]);
    h_lds[tid + 256] = f2bf(hin[tid + 256]);
    __syncthreads();

    v8f acc = {};
    for (int kt = 0; kt < HDIM; kt += 32) {
      v16bf a = load_a_frag(arows + kt, HDIM);
      v16bf h = load_b_frag_bcast(h_lds + kt);
      acc = __builtin_amdgcn_wmma_f32_16x16x32_bf16(false, a, false, h, (short)0, acc,
                                                    false, false);
    }
    if ((lane & 15) == 0) {                 // lanes 0/16 hold M=0..7 / 8..15 of col 0
      int mb = (lane >> 4) << 3;
      for (int r = 0; r < 8; ++r) r_lds[wave * 16 + mb + r] = acc[r];
    }
    __syncthreads();

    if (tid < 32) {
      int j = jbase + tid;
      const float* gx = gates + (long)t * G4;
      float iv = r_lds[tid]      + gx[j];
      float fv = r_lds[32 + tid] + gx[512  + j];
      float gv = r_lds[64 + tid] + gx[1024 + j];
      float ov = r_lds[96 + tid] + gx[1536 + j];
      iv = 1.f / (1.f + __expf(-iv));
      fv = 1.f / (1.f + __expf(-fv));
      gv = tanhf(gv);
      ov = 1.f / (1.f + __expf(-ov));
      c_reg = fv * c_reg + iv * gv;
      float hv = ov * tanhf(c_reg);
      hout[j] = hv;
      hs_bf[(long)t * HDIM + j] = f2bf(hv);
    }
    __syncthreads();
    if (tid == 0) {                         // device-scope grid barrier (16 resident blocks)
      __threadfence();
      __hip_atomic_fetch_add(sync, 1u, __ATOMIC_RELEASE, __HIP_MEMORY_SCOPE_AGENT);
      unsigned target = (unsigned)(t + 1) * gridDim.x;
      while (__hip_atomic_load(sync, __ATOMIC_ACQUIRE, __HIP_MEMORY_SCOPE_AGENT) < target)
        __builtin_amdgcn_s_sleep(1);
    }
    __syncthreads();
  }
}

// ---------------- output GEMM: async-LDS staged A tile + log_softmax -----------------
__global__ void out_kernel(const unsigned short* __restrict__ hs_bf,
                           const unsigned short* __restrict__ outw_bf,
                           const float* __restrict__ out_b, float* out) {
  __shared__ unsigned short atile[16 * HDIM] __attribute__((aligned(128)));  // 16 KB
  __shared__ float logits[16][PHONE];
  int tid  = threadIdx.x;
  int lane = tid & 31;
  int wave = tid >> 5;                      // 3 waves -> N tiles 0..2 (48 phones)
  int m0 = blockIdx.x * 16;
  int n0 = wave * 16;

  // stage hs[m0:m0+16, :] (16 KB) into LDS with async b128 copies
  {
    unsigned lds_base = lds_offset(&atile[0]);
    const char* gsrc = (const char*)(hs_bf + (long)m0 * HDIM);
    for (int c = tid; c < 1024; c += 96)
      async_ld_b128(lds_base + c * 16, gsrc + (long)c * 16);
    wait_asynccnt0();
  }
  __syncthreads();

  v8f acc = {};
  const unsigned short* bbase = outw_bf + (long)n0 * HDIM;
  for (int k0 = 0; k0 < HDIM; k0 += 32) {
    v16bf a = load_a_frag(atile + k0, HDIM);
    v16bf b = load_b_frag(bbase + k0, HDIM);
    acc = __builtin_amdgcn_wmma_f32_16x16x32_bf16(false, a, false, b, (short)0, acc,
                                                  false, false);
  }
  int col   = n0 + (lane & 15);
  int rbase = (lane >> 4) << 3;
  float bv = out_b[col];
  for (int r = 0; r < 8; ++r) logits[rbase + r][col] = acc[r] + bv;
  __syncthreads();
  if (tid < 16) {
    float mx = -1e30f;
    for (int p = 0; p < PHONE; ++p) mx = fmaxf(mx, logits[tid][p]);
    float s = 0.f;
    for (int p = 0; p < PHONE; ++p) s += __expf(logits[tid][p] - mx);
    float lse = mx + __logf(s);
    float* orow = out + (long)(m0 + tid) * PHONE;
    for (int p = 0; p < PHONE; ++p) orow[p] = logits[tid][p] - lse;
  }
}

extern "C" void kernel_launch(void* const* d_in, const int* in_sizes, int n_in,
                              void* d_out, int out_size, void* d_ws, size_t ws_size,
                              hipStream_t stream) {
  const float* input_seq = (const float*)d_in[0];
  const float* h0        = (const float*)d_in[1];
  const float* c0        = (const float*)d_in[2];
  const float* conv_w    = (const float*)d_in[3];
  const float* conv_b    = (const float*)d_in[4];
  const float* w_ih      = (const float*)d_in[5];
  const float* w_hh      = (const float*)d_in[6];
  const float* b_ih      = (const float*)d_in[7];
  const float* b_hh      = (const float*)d_in[8];
  const float* out_w     = (const float*)d_in[9];
  const float* out_b     = (const float*)d_in[10];

  char* ws = (char*)d_ws;
  size_t off = 0;
  auto alloc = [&](size_t bytes) -> void* {
    off = (off + 255) & ~(size_t)255;
    void* p = ws + off;
    off += bytes;
    return p;
  };
  unsigned short* xin     = (unsigned short*)alloc((size_t)T_FRAMES * KPAD * 2);
  unsigned short* wih_bf  = (unsigned short*)alloc((size_t)G4 * KPAD * 2);
  unsigned short* whh_bf  = (unsigned short*)alloc((size_t)G4 * HDIM * 2);
  unsigned short* outw_bf = (unsigned short*)alloc((size_t)PHONE * HDIM * 2);
  float*          bias    = (float*)alloc((size_t)G4 * 4);
  float*          hbuf    = (float*)alloc((size_t)2 * HDIM * 4);
  unsigned short* hs_bf   = (unsigned short*)alloc((size_t)T_FRAMES * HDIM * 2);
  float*          gates   = (float*)alloc((size_t)T_FRAMES * G4 * 4);
  unsigned*       sync    = (unsigned*)alloc(256);

  long prep_n = (long)G4 * KPAD + (long)G4 * HDIM + (long)PHONE * HDIM + G4 + HDIM + 1;
  int prep_blocks = (int)((prep_n + 255) / 256);

  prep_kernel<<<prep_blocks, 256, 0, stream>>>(w_ih, w_hh, out_w, b_ih, b_hh, h0,
                                               wih_bf, whh_bf, outw_bf, bias, hbuf, sync);
  feat_kernel<<<T_FRAMES, 256, 0, stream>>>(input_seq, conv_w, conv_b, xin);
  gates_gemm_kernel<<<T_FRAMES, 256, 0, stream>>>(xin, wih_bf, bias, gates);
  lstm_kernel<<<16, 256, 0, stream>>>(gates, whh_bf, c0, hbuf, hs_bf, sync);
  out_kernel<<<T_FRAMES / 16, 96, 0, stream>>>(hs_bf, outw_bf, out_b, (float*)d_out);
}
